// SelfAttention_23278722744838
// MI455X (gfx1250) — matrix operand order repaired and checked by reading, not compiled, not run on hardware
//
#include <hip/hip_runtime.h>
#include <stdint.h>

// ---------------------------------------------------------------------------
// Types for CDNA5 WMMA (wave32): A/B = 16 bf16 per lane, C/D = 8 f32 per lane
// ---------------------------------------------------------------------------
typedef __attribute__((ext_vector_type(16))) __bf16 bf16x16;
typedef __attribute__((ext_vector_type(8)))  float  f32x8;
typedef __attribute__((ext_vector_type(4)))  int    v4i;
typedef unsigned short u16;

union BV {                 // operand assembly container
    bf16x16 v;
    uint4   q[2];          // two 16B halves
    u16     u[16];
};

__device__ __forceinline__ u16 f2bf(float f) {   // round-to-nearest-even f32->bf16
    uint32_t x = __float_as_uint(f);
    uint32_t r = (x + 0x7FFFu + ((x >> 16) & 1u)) >> 16;
    return (u16)r;
}

#define BATCH 4
#define CCH   512          // channels
#define NTOK  2304         // 48*48
#define NH    8            // heads
#define DH    64           // dim head
#define NGRP  32
#define CPG   16           // channels per group
#define WMMA_BF16(A,B,C) __builtin_amdgcn_wmma_f32_16x16x32_bf16(false,(A),false,(B),(short)0,(C),false,false)

// ---------------------------------------------------------------------------
// CDNA5 async global->LDS copy (ASYNCcnt path), builtin if present, asm else.
// Builtin signature (from clang diagnostic): param0 = v4i addrspace(1)*,
// so: (global src, lds dst, imm offset, imm cpol).
// ---------------------------------------------------------------------------
typedef __attribute__((address_space(1))) v4i g_v4i;
typedef __attribute__((address_space(3))) v4i l_v4i;

__device__ __forceinline__ void async_copy16(void* lds_ptr, const void* gptr) {
#if __has_builtin(__builtin_amdgcn_global_load_async_to_lds_b128)
    __builtin_amdgcn_global_load_async_to_lds_b128(
        (g_v4i*)(void*)(uintptr_t)gptr, (l_v4i*)lds_ptr, 0, 0);
#else
    unsigned loff = (unsigned)(uintptr_t)(__attribute__((address_space(3))) void*)lds_ptr;
    asm volatile("global_load_async_to_lds_b128 %0, %1, off"
                 :: "v"(loff), "v"(gptr) : "memory");
#endif
}

#if __has_builtin(__builtin_amdgcn_s_wait_asynccnt)
#define WAIT_ASYNC(n) __builtin_amdgcn_s_wait_asynccnt(n)
#else
#define WAIT_ASYNC(n) asm volatile("s_wait_asynccnt %0" :: "i"(n) : "memory")
#endif

// ---------------------------------------------------------------------------
// Operand tile bundle for the dense GEMMs (A + 4 B tiles for one K step)
// ---------------------------------------------------------------------------
struct GTiles { BV a, b0, b1, b2, b3; };

__device__ __forceinline__ GTiles load_gtiles(const u16* arow,
                                              const u16* b0p, const u16* b1p,
                                              const u16* b2p, const u16* b3p,
                                              int c, int hi) {
    GTiles t;
    t.a.q[0] = *(const uint4*)(arow + c + hi * 16);         // K = c..c+7 (+16 for hi half)
    t.a.q[1] = *(const uint4*)(arow + c + hi * 16 + 8);     // K = c+8.. (+16 for hi half)
    const int off = c + hi * 16;                            // B: 16-K chunk, lo/hi halves
    t.b0.q[0] = *(const uint4*)(b0p + off); t.b0.q[1] = *(const uint4*)(b0p + off + 8);
    t.b1.q[0] = *(const uint4*)(b1p + off); t.b1.q[1] = *(const uint4*)(b1p + off + 8);
    t.b2.q[0] = *(const uint4*)(b2p + off); t.b2.q[1] = *(const uint4*)(b2p + off + 8);
    t.b3.q[0] = *(const uint4*)(b3p + off); t.b3.q[1] = *(const uint4*)(b3p + off + 8);
    return t;
}

// ---------------------------------------------------------------------------
// 1) GroupNorm statistics: one block per (b, group); 16*2304 contiguous floats
// ---------------------------------------------------------------------------
__global__ void __launch_bounds__(256) gn_stats_kernel(const float* __restrict__ x,
                                                       float* __restrict__ stats) {
    const int b = blockIdx.x >> 5;
    const int g = blockIdx.x & 31;
    const float* p = x + ((size_t)(b * CCH + g * CPG)) * NTOK;
    float s = 0.f, ss = 0.f;
    for (int i = threadIdx.x; i < CPG * NTOK; i += 256) {
        float v = p[i];
        s += v; ss += v * v;
    }
    #pragma unroll
    for (int m = 1; m < 32; m <<= 1) { s += __shfl_xor(s, m); ss += __shfl_xor(ss, m); }
    __shared__ float as[8], bs[8];
    if ((threadIdx.x & 31) == 0) { as[threadIdx.x >> 5] = s; bs[threadIdx.x >> 5] = ss; }
    __syncthreads();
    if (threadIdx.x == 0) {
        float S = 0.f, SS = 0.f;
        #pragma unroll
        for (int i = 0; i < 8; ++i) { S += as[i]; SS += bs[i]; }
        const float inv_n = 1.0f / (float)(CPG * NTOK);
        float mean = S * inv_n;
        float var  = SS * inv_n - mean * mean;
        stats[2 * blockIdx.x + 0] = mean;
        stats[2 * blockIdx.x + 1] = rsqrtf(var + 1e-5f);
    }
}

// ---------------------------------------------------------------------------
// 2) Apply GN, convert to bf16, transpose to xn[b, n, c]  (c contiguous = K)
// ---------------------------------------------------------------------------
__global__ void __launch_bounds__(256) gn_apply_kernel(const float* __restrict__ x,
                                                       const float* __restrict__ stats,
                                                       const float* __restrict__ gamma,
                                                       const float* __restrict__ beta,
                                                       u16* __restrict__ xn) {
    size_t idx = (size_t)blockIdx.x * 256 + threadIdx.x;   // [b, c, n] order (coalesced read)
    int b = (int)(idx / ((size_t)CCH * NTOK));
    int rem = (int)(idx % ((size_t)CCH * NTOK));
    int c = rem / NTOK;
    int n = rem % NTOK;
    int g = c >> 4;
    float mean = stats[2 * (b * NGRP + g) + 0];
    float rstd = stats[2 * (b * NGRP + g) + 1];
    float v = (x[idx] - mean) * rstd * gamma[c] + beta[c];
    xn[((size_t)b * NTOK + n) * CCH + c] = f2bf(v);
}

// ---------------------------------------------------------------------------
// 3) Weight conversion fp32 -> bf16 (w_qkv then w_out)
// ---------------------------------------------------------------------------
__global__ void __launch_bounds__(256) wconv_kernel(const float* __restrict__ wqkv,
                                                    const float* __restrict__ wout,
                                                    u16* __restrict__ wqb,
                                                    u16* __restrict__ wob) {
    int i = blockIdx.x * 256 + threadIdx.x;
    const int NQ = 3 * CCH * CCH;       // 786432
    if (i < NQ) wqb[i] = f2bf(wqkv[i]);
    else        wob[i - NQ] = f2bf(wout[i - NQ]);
}

// ---------------------------------------------------------------------------
// 4) QKV GEMM: each wave owns a 16(M=o) x 64(N=n) strip -> 4 accumulators,
//    A operand reused across 4 WMMAs per K step (K=512). Software-pipelined:
//    K-step c+32 operand tiles are fetched while step c's WMMAs execute.
//    q,k stored [b,h,n,d]; v stored transposed [b,h,d,n]
// ---------------------------------------------------------------------------
__global__ void __launch_bounds__(256) qkv_gemm_kernel(const u16* __restrict__ wq,
                                                       const u16* __restrict__ xn,
                                                       u16* __restrict__ qb,
                                                       u16* __restrict__ kb,
                                                       u16* __restrict__ vb) {
    const int lane = threadIdx.x & 31;
    const int hi = lane >> 4, lo = lane & 15;
    int gw = blockIdx.x * 8 + (threadIdx.x >> 5);
    int nt = gw % 36;                 // 36 strips of 64 columns
    int ot = (gw / 36) % 96;
    int b  = gw / (36 * 96);

    const u16* arow = wq + (size_t)(ot * 16 + lo) * CCH;              // A: lane = M row
    const u16* bcol0 = xn + ((size_t)b * NTOK + nt * 64 +  0 + lo) * CCH;
    const u16* bcol1 = xn + ((size_t)b * NTOK + nt * 64 + 16 + lo) * CCH;
    const u16* bcol2 = xn + ((size_t)b * NTOK + nt * 64 + 32 + lo) * CCH;
    const u16* bcol3 = xn + ((size_t)b * NTOK + nt * 64 + 48 + lo) * CCH;

    f32x8 acc0 = {}, acc1 = {}, acc2 = {}, acc3 = {};
    GTiles cur = load_gtiles(arow, bcol0, bcol1, bcol2, bcol3, 0, hi);
    for (int c = 32; c < CCH; c += 32) {
        GTiles nxt = load_gtiles(arow, bcol0, bcol1, bcol2, bcol3, c, hi);
        acc0 = WMMA_BF16(cur.a.v, cur.b0.v, acc0);
        acc1 = WMMA_BF16(cur.a.v, cur.b1.v, acc1);
        acc2 = WMMA_BF16(cur.a.v, cur.b2.v, acc2);
        acc3 = WMMA_BF16(cur.a.v, cur.b3.v, acc3);
        cur = nxt;
    }
    acc0 = WMMA_BF16(cur.a.v, cur.b0.v, acc0);
    acc1 = WMMA_BF16(cur.a.v, cur.b1.v, acc1);
    acc2 = WMMA_BF16(cur.a.v, cur.b2.v, acc2);
    acc3 = WMMA_BF16(cur.a.v, cur.b3.v, acc3);

    f32x8 accs[4] = {acc0, acc1, acc2, acc3};
    #pragma unroll
    for (int t = 0; t < 4; ++t) {
        #pragma unroll
        for (int r = 0; r < 8; ++r) {                         // C layout: M = r + 8*hi, N = lo
            int o = ot * 16 + r + 8 * hi;
            int n = nt * 64 + t * 16 + lo;
            int which = o >> 9;                               // 0:q 1:k 2:v (tiles never straddle)
            int oo = o & 511;
            int h = oo >> 6, d = oo & 63;
            u16 val = f2bf(accs[t][r]);
            if (which == 0)      qb[(((size_t)b * NH + h) * NTOK + n) * DH + d] = val;
            else if (which == 1) kb[(((size_t)b * NH + h) * NTOK + n) * DH + d] = val;
            else                 vb[(((size_t)b * NH + h) * DH + d) * NTOK + n] = val;
        }
    }
}

// ---------------------------------------------------------------------------
// 5) Flash attention with double-buffered async global->LDS staging of K/V.
//    All 8 waves share (b,h): each 32-key chunk (K tile 4KB + V tile 4KB) is
//    fetched once per block via global_load_async_to_lds_b128 (one 16B piece
//    per thread), drained with s_wait_asynccnt, and consumed from LDS.
// ---------------------------------------------------------------------------
__global__ void __launch_bounds__(256) attention_kernel(const u16* __restrict__ qb,
                                                        const u16* __restrict__ kb,
                                                        const u16* __restrict__ vb,
                                                        u16* __restrict__ ao) {
    __shared__ u16 ldsK[2][32 * DH];   // [buf][key row][d]     4KB each
    __shared__ u16 ldsV[2][DH * 32];   // [buf][d][key]         4KB each
    __shared__ u16 ldsP[8][16 * 32];   // per-wave P tile       1KB each
    const int tid  = threadIdx.x;
    const int lane = tid & 31;
    const int w    = tid >> 5;
    const int hi = lane >> 4, lo = lane & 15;
    int bh = blockIdx.x / 18;
    int qt = (blockIdx.x % 18) * 8 + w;              // query tile (16 rows)
    int b = bh >> 3, h = bh & 7;

    const u16* qh = qb + ((size_t)b * NH + h) * NTOK * DH;
    const u16* kh = kb + ((size_t)b * NH + h) * NTOK * DH;
    const u16* vh = vb + ((size_t)b * NH + h) * DH * NTOK;

    // cooperative async stage of one 32-key chunk of K and V
    const int krow = tid >> 3, kpart = tid & 7;      // 32 rows x 8 pieces (128B/row)
    const int vd   = tid >> 2, vpart = tid & 3;      // 64 d-rows x 4 pieces (64B/row)
    auto issue = [&](int buf, int m0) {
        async_copy16(&ldsK[buf][krow * DH + kpart * 8],
                     kh + (size_t)(m0 + krow) * DH + kpart * 8);
        async_copy16(&ldsV[buf][vd * 32 + vpart * 8],
                     vh + (size_t)vd * NTOK + m0 + vpart * 8);
    };
    issue(0, 0);

    const u16* qr = qh + (size_t)(qt * 16 + lo) * DH;    // A: lane = query row
    BV aq0, aq1;
    aq0.q[0] = *(const uint4*)(qr + hi * 16);
    aq0.q[1] = *(const uint4*)(qr + hi * 16 + 8);
    aq1.q[0] = *(const uint4*)(qr + 32 + hi * 16);
    aq1.q[1] = *(const uint4*)(qr + 32 + hi * 16 + 8);

    float mrun[8], lrun[8];
    #pragma unroll
    for (int r = 0; r < 8; ++r) { mrun[r] = -3.0e38f; lrun[r] = 0.f; }
    f32x8 acc0 = {}, acc1 = {}, acc2 = {}, acc3 = {};
    u16* pl = &ldsP[w][0];
    const float sc = 0.125f;                        // 1/sqrt(64)

    for (int it = 0, m0 = 0; m0 < NTOK; ++it, m0 += 32) {
        const int cur = it & 1;
        if (m0 + 32 < NTOK) { issue(cur ^ 1, m0 + 32); WAIT_ASYNC(2); }
        else                { WAIT_ASYNC(0); }
        __syncthreads();                            // chunk `cur` visible to all waves

        // S tiles from LDS-resident K: B operand lane = key col, 16-K(d) chunk.
        const u16* kbase = &ldsK[cur][0];
        BV bk0, bk1, bk2, bk3;
        bk0.q[0] = *(const uint4*)(kbase + lo * DH + hi * 16);
        bk0.q[1] = *(const uint4*)(kbase + lo * DH + hi * 16 + 8);
        bk1.q[0] = *(const uint4*)(kbase + lo * DH + 32 + hi * 16);
        bk1.q[1] = *(const uint4*)(kbase + lo * DH + 32 + hi * 16 + 8);
        bk2.q[0] = *(const uint4*)(kbase + (16 + lo) * DH + hi * 16);
        bk2.q[1] = *(const uint4*)(kbase + (16 + lo) * DH + hi * 16 + 8);
        bk3.q[0] = *(const uint4*)(kbase + (16 + lo) * DH + 32 + hi * 16);
        bk3.q[1] = *(const uint4*)(kbase + (16 + lo) * DH + 32 + hi * 16 + 8);
        f32x8 S0 = {}, S1 = {};
        S0 = WMMA_BF16(aq0.v, bk0.v, S0);
        S0 = WMMA_BF16(aq1.v, bk1.v, S0);
        S1 = WMMA_BF16(aq0.v, bk2.v, S1);
        S1 = WMMA_BF16(aq1.v, bk3.v, S1);

        #pragma unroll
        for (int r = 0; r < 8; ++r) {               // row = r + 8*hi, striped over 16 lanes
            float a = S0[r] * sc, c = S1[r] * sc;
            float rmax = fmaxf(a, c);
            rmax = fmaxf(rmax, __shfl_xor(rmax, 1));
            rmax = fmaxf(rmax, __shfl_xor(rmax, 2));
            rmax = fmaxf(rmax, __shfl_xor(rmax, 4));
            rmax = fmaxf(rmax, __shfl_xor(rmax, 8));
            float mnew = fmaxf(mrun[r], rmax);
            float p0 = __expf(a - mnew);
            float p1 = __expf(c - mnew);
            float rs = p0 + p1;
            rs += __shfl_xor(rs, 1);
            rs += __shfl_xor(rs, 2);
            rs += __shfl_xor(rs, 4);
            rs += __shfl_xor(rs, 8);
            float alpha = __expf(mrun[r] - mnew);
            lrun[r] = lrun[r] * alpha + rs;
            mrun[r] = mnew;
            acc0[r] *= alpha; acc1[r] *= alpha; acc2[r] *= alpha; acc3[r] *= alpha;
            int row = r + 8 * hi;
            pl[row * 32 + lo]      = f2bf(p0);
            pl[row * 32 + 16 + lo] = f2bf(p1);
        }
        __syncthreads();                            // P tile ready

        BV ap;                                      // P back in A layout (lane = row)
    ap.q[0] = *(const uint4*)(pl + lo * 32 + hi * 16);
        ap.q[1] = *(const uint4*)(pl + lo * 32 + hi * 16 + 8);

        const u16* vbase = &ldsV[cur][0];           // B: lane = d col, contiguous 16 keys
        BV bv0, bv1, bv2, bv3;
        bv0.q[0] = *(const uint4*)(vbase + ( 0 + lo) * 32 + hi * 16);
        bv0.q[1] = *(const uint4*)(vbase + ( 0 + lo) * 32 + hi * 16 + 8);
        bv1.q[0] = *(const uint4*)(vbase + (16 + lo) * 32 + hi * 16);
        bv1.q[1] = *(const uint4*)(vbase + (16 + lo) * 32 + hi * 16 + 8);
        bv2.q[0] = *(const uint4*)(vbase + (32 + lo) * 32 + hi * 16);
        bv2.q[1] = *(const uint4*)(vbase + (32 + lo) * 32 + hi * 16 + 8);
        bv3.q[0] = *(const uint4*)(vbase + (48 + lo) * 32 + hi * 16);
        bv3.q[1] = *(const uint4*)(vbase + (48 + lo) * 32 + hi * 16 + 8);
        acc0 = WMMA_BF16(ap.v, bv0.v, acc0);
        acc1 = WMMA_BF16(ap.v, bv1.v, acc1);
        acc2 = WMMA_BF16(ap.v, bv2.v, acc2);
        acc3 = WMMA_BF16(ap.v, bv3.v, acc3);

        __syncthreads();                            // all reads done before buffer reuse
    }

    #pragma unroll
    for (int r = 0; r < 8; ++r) {
        float inv = 1.0f / lrun[r];
        int n = qt * 16 + r + 8 * hi;
        u16* orow = ao + ((size_t)b * NTOK + n) * CCH + h * DH;   // ao[b, n, h*64+d]
        orow[0 * 16 + lo] = f2bf(acc0[r] * inv);
        orow[1 * 16 + lo] = f2bf(acc1[r] * inv);
        orow[2 * 16 + lo] = f2bf(acc2[r] * inv);
        orow[3 * 16 + lo] = f2bf(acc3[r] * inv);
    }
}

// ---------------------------------------------------------------------------
// 6) Output projection: 16(M=c) x 64(N=n) strip per wave, software-pipelined.
//    out[b,c,n] = w_out[c,:] . ao[b,n,:] + b_out[c]
// ---------------------------------------------------------------------------
__global__ void __launch_bounds__(256) outproj_kernel(const u16* __restrict__ wo,
                                                      const u16* __restrict__ ao,
                                                      const float* __restrict__ bout,
                                                      float* __restrict__ out) {
    const int lane = threadIdx.x & 31;
    const int hi = lane >> 4, lo = lane & 15;
    int gw = blockIdx.x * 8 + (threadIdx.x >> 5);
    int nt = gw % 36;
    int ct = (gw / 36) % 32;
    int b  = gw / (36 * 32);

    const u16* arow = wo + (size_t)(ct * 16 + lo) * CCH;
    const u16* bcol0 = ao + ((size_t)b * NTOK + nt * 64 +  0 + lo) * CCH;
    const u16* bcol1 = ao + ((size_t)b * NTOK + nt * 64 + 16 + lo) * CCH;
    const u16* bcol2 = ao + ((size_t)b * NTOK + nt * 64 + 32 + lo) * CCH;
    const u16* bcol3 = ao + ((size_t)b * NTOK + nt * 64 + 48 + lo) * CCH;

    f32x8 acc0 = {}, acc1 = {}, acc2 = {}, acc3 = {};
    GTiles cur = load_gtiles(arow, bcol0, bcol1, bcol2, bcol3, 0, hi);
    for (int c = 32; c < CCH; c += 32) {
        GTiles nxt = load_gtiles(arow, bcol0, bcol1, bcol2, bcol3, c, hi);
        acc0 = WMMA_BF16(cur.a.v, cur.b0.v, acc0);
        acc1 = WMMA_BF16(cur.a.v, cur.b1.v, acc1);
        acc2 = WMMA_BF16(cur.a.v, cur.b2.v, acc2);
        acc3 = WMMA_BF16(cur.a.v, cur.b3.v, acc3);
        cur = nxt;
    }
    acc0 = WMMA_BF16(cur.a.v, cur.b0.v, acc0);
    acc1 = WMMA_BF16(cur.a.v, cur.b1.v, acc1);
    acc2 = WMMA_BF16(cur.a.v, cur.b2.v, acc2);
    acc3 = WMMA_BF16(cur.a.v, cur.b3.v, acc3);

    f32x8 accs[4] = {acc0, acc1, acc2, acc3};
    #pragma unroll
    for (int t = 0; t < 4; ++t) {
        #pragma unroll
        for (int r = 0; r < 8; ++r) {
            int crow = ct * 16 + r + 8 * hi;
            int n = nt * 64 + t * 16 + lo;
            out[((size_t)b * CCH + crow) * NTOK + n] = accs[t][r] + bout[crow];
        }
    }
}

// ---------------------------------------------------------------------------
// Host orchestration
// ---------------------------------------------------------------------------
extern "C" void kernel_launch(void* const* d_in, const int* in_sizes, int n_in,
                              void* d_out, int out_size, void* d_ws, size_t ws_size,
                              hipStream_t stream) {
    (void)in_sizes; (void)n_in; (void)out_size; (void)ws_size;
    const float* x     = (const float*)d_in[0];
    const float* gamma = (const float*)d_in[1];
    const float* beta  = (const float*)d_in[2];
    const float* wqkv  = (const float*)d_in[3];
    const float* wout  = (const float*)d_in[4];
    const float* bout  = (const float*)d_in[5];
    float* out = (float*)d_out;

    const size_t TENS = (size_t)BATCH * NTOK * CCH;      // 4,718,592 elements
    char* ws = (char*)d_ws;
    float* stats = (float*)ws;                            // 128 * 2 f32
    u16* xn  = (u16*)(ws + 4096);
    u16* qb  = xn + TENS;
    u16* kb  = qb + TENS;
    u16* vb  = kb + TENS;
    u16* ao  = vb + TENS;
    u16* wqb = ao + TENS;
    u16* wob = wqb + (size_t)3 * CCH * CCH;

    gn_stats_kernel<<<BATCH * NGRP, 256, 0, stream>>>(x, stats);
    gn_apply_kernel<<<(int)(TENS / 256), 256, 0, stream>>>(x, stats, gamma, beta, xn);
    wconv_kernel<<<(3 * CCH * CCH + CCH * CCH) / 256, 256, 0, stream>>>(wqkv, wout, wqb, wob);
    // 4 * 96 * 36 wave-strips / 8 waves per block
    qkv_gemm_kernel<<<(BATCH * 96 * 36) / 8, 256, 0, stream>>>(wqb, xn, qb, kb, vb);
    // 4 * 8 heads * 18 blocks (8 q-tiles each)
    attention_kernel<<<BATCH * NH * 18, 256, 0, stream>>>(qb, kb, vb, ao);
    // 4 * 32 * 36 wave-strips / 8
    outproj_kernel<<<(BATCH * 32 * 36) / 8, 256, 0, stream>>>(wob, ao, bout, out);
}
